// DGCNN_cls_16509854285875
// MI455X (gfx1250) — matrix operand order repaired and checked by reading, not compiled, not run on hardware
//
#include <hip/hip_runtime.h>
#include <math.h>

typedef __attribute__((ext_vector_type(16))) _Float16 v16h;
typedef __attribute__((ext_vector_type(8)))  float    v8f;

#define NEGS 0.2f
#define EPSV 1e-5f

__device__ __forceinline__ float leaky(float v) { return v >= 0.f ? v : NEGS * v; }

// ---------------------------------------------------------------------------
// kNN: one block per (query m, batch b). Distances in LDS, iterative top-k
// with lowest-index tie break (matches jax.lax.top_k ordering).
// ---------------------------------------------------------------------------
__global__ void knn_kernel(const float* __restrict__ q, const float* __restrict__ ref,
                           int C, int qN, int rN, int M, int Nref, int k,
                           int qBs, int rBs, int* __restrict__ idxout) {
    extern __shared__ char kmem[];
    float* dist = (float*)kmem;           // Nref
    float* qv   = dist + Nref;            // C
    float* bv   = qv + C;                 // 128
    int*   bi   = (int*)(bv + 128);       // 128
    const int b = blockIdx.y, m = blockIdx.x;
    const float* qb = q   + (size_t)b * qBs;
    const float* rb = ref + (size_t)b * rBs;
    for (int c = threadIdx.x; c < C; c += blockDim.x) qv[c] = qb[(size_t)c * qN + m];
    __syncthreads();
    for (int j = threadIdx.x; j < Nref; j += blockDim.x) {
        float d = 0.f;
        for (int c = 0; c < C; ++c) { float t = qv[c] - rb[(size_t)c * rN + j]; d += t * t; }
        dist[j] = -d;
    }
    __syncthreads();
    int* outp = idxout + ((size_t)b * M + m) * k;
    for (int t = 0; t < k; ++t) {
        float best = -3.4e38f; int bidx = 0x7fffffff;
        for (int j = threadIdx.x; j < Nref; j += blockDim.x) {
            float v = dist[j];
            if (v > best || (v == best && j < bidx)) { best = v; bidx = j; }
        }
        bv[threadIdx.x] = best; bi[threadIdx.x] = bidx;
        __syncthreads();
        for (int s = blockDim.x >> 1; s > 0; s >>= 1) {
            if ((int)threadIdx.x < s) {
                float ov = bv[threadIdx.x + s]; int oi = bi[threadIdx.x + s];
                if (ov > bv[threadIdx.x] || (ov == bv[threadIdx.x] && oi < bi[threadIdx.x])) {
                    bv[threadIdx.x] = ov; bi[threadIdx.x] = oi;
                }
            }
            __syncthreads();
        }
        if (threadIdx.x == 0) { outp[t] = bi[0]; dist[bi[0]] = -3.4e38f; }
        __syncthreads();
    }
}

// ---------------------------------------------------------------------------
// Pack weight W (O x Ctrue, f32, row major) into WMMA A-fragment order (f16).
// Fragment index t = (kc*numOT + ot)*32 + lane ; 16 halves per t.
// ---------------------------------------------------------------------------
__global__ void packW_kernel(const float* __restrict__ W, _Float16* __restrict__ Wp,
                             int O, int Ctrue, int Cpad, int total) {
    int t = blockIdx.x * blockDim.x + threadIdx.x;
    if (t >= total) return;
    int lane = t & 31; int rest = t >> 5;
    int numOT = O >> 4;
    int ot = rest % numOT; int kc = rest / numOT;
    int row = ot * 16 + (lane & 15);
    for (int e = 0; e < 16; ++e) {
        int koff = ((lane < 16) ? 0 : 16) + e;
        int kk = kc * 32 + koff;
        float v = (kk < Ctrue) ? W[(size_t)row * Ctrue + kk] : 0.f;
        Wp[(size_t)t * 16 + e] = (_Float16)v;
    }
}

// ---------------------------------------------------------------------------
// Pack activations (and optional channel-concat second source) into f16
// column-major-by-point layout: out[b][n][c], c padded to Cpad (mult of 32).
// ---------------------------------------------------------------------------
__global__ void packX_kernel(const float* __restrict__ s1, int C1, int s1N,
                             const float* __restrict__ s2, int C2,
                             _Float16* __restrict__ out, int Ncols, int Cpad, int total) {
    int t = blockIdx.x * blockDim.x + threadIdx.x;
    if (t >= total) return;
    int c = t % Cpad; int n = (t / Cpad) % Ncols; int b = t / (Cpad * Ncols);
    float v = 0.f;
    if (c < C1)               v = s1[((size_t)b * C1 + c) * s1N + n];
    else if (c < C1 + C2)     v = s2[((size_t)b * C2 + (c - C1)) * Ncols + n];
    out[t] = (_Float16)v;
}

// ---------------------------------------------------------------------------
// Fused EdgeConv: gather K neighbors, build [f-xc ; xc] tile (2C x COLS) in
// LDS (f16), GEMM with WMMA f16->f32, accumulate per-channel sum/sumsq
// (exact BN stats over B,N,K) and per-point max over k.
// ---------------------------------------------------------------------------
template <int Kk, int COLS>
__global__ void edgeconv_gemm(const float* __restrict__ x, int Cin, int Nn, int Cpad2,
                              const int* __restrict__ idx, const _Float16* __restrict__ Wp,
                              int O, float* __restrict__ ymax,
                              float* __restrict__ chsum, float* __restrict__ chsq) {
    constexpr int NT = COLS / Kk;
    extern __shared__ char smem[];
    _Float16* E = (_Float16*)smem;                // COLS * Cpad2 halves
    int* sidx = (int*)(E + (size_t)COLS * Cpad2); // COLS ints
    const int b = blockIdx.y;
    const int p0 = blockIdx.x * NT;
    const int tid = threadIdx.x;
    if (tid < COLS) {
        int p = tid / Kk, kk2 = tid % Kk;
        sidx[tid] = idx[((size_t)b * Nn + p0 + p) * Kk + kk2];
    }
    __syncthreads();
    const float* xb = x + (size_t)b * Cin * Nn;
    for (int t = tid; t < COLS * Cpad2; t += blockDim.x) {
        int col = t & (COLS - 1); int c = t / COLS;
        float v = 0.f;
        if (c < 2 * Cin) {
            int p = col / Kk; int nc = p0 + p;
            if (c < Cin) { int j = sidx[col]; v = xb[(size_t)c * Nn + j] - xb[(size_t)c * Nn + nc]; }
            else         { v = xb[(size_t)(c - Cin) * Nn + nc]; }
        }
        E[(size_t)col * Cpad2 + c] = (_Float16)v;
    }
    __syncthreads();
    const int wid = tid >> 5, lane = tid & 31;
    const int numOT = O >> 4;
    const int nloc = lane & 15;
    const _Float16* e0 = &E[(size_t)nloc * Cpad2 + ((lane < 16) ? 0 : 16)];
    const _Float16* e1 = &E[(size_t)((nloc + 16) & (COLS - 1)) * Cpad2 + ((lane < 16) ? 0 : 16)];
    for (int ot = wid; ot < numOT; ot += 4) {
        v8f a0 = {}; v8f a1 = {};
        for (int kc = 0; kc < (Cpad2 >> 5); ++kc) {
            v16h af = *(const v16h*)&Wp[(((size_t)kc * numOT + ot) * 32 + lane) * 16];
            v16h b0 = *(const v16h*)&e0[(size_t)kc * 32];
            a0 = __builtin_amdgcn_wmma_f32_16x16x32_f16(false, af, false, b0, (short)0, a0, false, false);
            if (COLS == 32) {
                v16h b1 = *(const v16h*)&e1[(size_t)kc * 32];
                a1 = __builtin_amdgcn_wmma_f32_16x16x32_f16(false, af, false, b1, (short)0, a1, false, false);
            }
        }
        constexpr int SEG = (Kk < 16) ? Kk : 16;
        #pragma unroll
        for (int i = 0; i < 8; ++i) {
            int o = ot * 16 + ((lane < 16) ? i : 8 + i);
            float v0 = a0[i];
            float v1 = (COLS == 32) ? a1[i] : 0.f;
            float rs = v0 + v1, rq = v0 * v0 + v1 * v1;
            #pragma unroll
            for (int mm = 8; mm >= 1; mm >>= 1) {
                rs += __shfl_xor(rs, mm, 32);
                rq += __shfl_xor(rq, mm, 32);
            }
            if (nloc == 0) { atomicAdd(&chsum[o], rs); atomicAdd(&chsq[o], rq); }
            float m0 = (Kk == 32) ? fmaxf(v0, v1) : v0;
            float m1 = v1;
            #pragma unroll
            for (int mm = SEG / 2; mm >= 1; mm >>= 1) {
                m0 = fmaxf(m0, __shfl_xor(m0, mm, 32));
                if (COLS == 32 && Kk < 32) m1 = fmaxf(m1, __shfl_xor(m1, mm, 32));
            }
            if ((nloc & (SEG - 1)) == 0) {
                int pl = nloc / SEG;
                ymax[((size_t)b * O + o) * Nn + p0 + pl] = m0;
                if (COLS == 32 && Kk < 32)
                    ymax[((size_t)b * O + o) * Nn + p0 + (16 / SEG) + pl] = m1;
            }
        }
    }
}

// ---------------------------------------------------------------------------
// Pointwise conv (c1) GEMM: y[b,o,n] = sum_c W[o,c]*x[b,c,n] via WMMA,
// with per-channel sum/sumsq accumulation for BN over (B,N).
// ---------------------------------------------------------------------------
__global__ void c1_gemm(const _Float16* __restrict__ xT, const _Float16* __restrict__ Wp,
                        float* __restrict__ yraw, float* __restrict__ chsum,
                        float* __restrict__ chsq, int Cpad, int O, int Nc) {
    const int b = blockIdx.y;
    const int ct = blockIdx.x;
    const int wid = threadIdx.x >> 5, lane = threadIdx.x & 31;
    const int numOT = O >> 4;
    const int nloc = lane & 15;
    const _Float16* xb = xT + ((size_t)b * Nc + ct * 16 + nloc) * Cpad + ((lane < 16) ? 0 : 16);
    for (int ot = wid; ot < numOT; ot += 4) {
        v8f acc = {};
        for (int kc = 0; kc < (Cpad >> 5); ++kc) {
            v16h a = *(const v16h*)&Wp[(((size_t)kc * numOT + ot) * 32 + lane) * 16];
            v16h bb = *(const v16h*)&xb[(size_t)kc * 32];
            acc = __builtin_amdgcn_wmma_f32_16x16x32_f16(false, a, false, bb, (short)0, acc, false, false);
        }
        #pragma unroll
        for (int i = 0; i < 8; ++i) {
            int o = ot * 16 + ((lane < 16) ? i : 8 + i);
            float v = acc[i];
            yraw[((size_t)b * O + o) * Nc + ct * 16 + nloc] = v;
            float rs = v, rq = v * v;
            #pragma unroll
            for (int mm = 8; mm >= 1; mm >>= 1) {
                rs += __shfl_xor(rs, mm, 32);
                rq += __shfl_xor(rq, mm, 32);
            }
            if (nloc == 0) { atomicAdd(&chsum[o], rs); atomicAdd(&chsq[o], rq); }
        }
    }
}

// ---------------------------------------------------------------------------
// BN(channel) + leaky (+ optional residual-then-leaky). In-place safe.
// ---------------------------------------------------------------------------
__global__ void finalize_kernel(const float* __restrict__ y, float* __restrict__ out,
                                const float* __restrict__ chsum, const float* __restrict__ chsq,
                                const float* __restrict__ g, const float* __restrict__ bb,
                                const float* __restrict__ res, int O, int Nc,
                                float invcount, int total) {
    int t = blockIdx.x * blockDim.x + threadIdx.x;
    if (t >= total) return;
    int o = (t / Nc) % O;
    float mean = chsum[o] * invcount;
    float var  = chsq[o] * invcount - mean * mean;
    float sc   = g[o] * rsqrtf(var + EPSV);
    float v = leaky((y[t] - mean) * sc + bb[o]);
    if (res) v = leaky(v + res[t]);
    out[t] = v;
}

// aggregate(): gather features at idx and take max over k
__global__ void gather_max_kernel(const float* __restrict__ feat, const int* __restrict__ idx,
                                  float* __restrict__ out, int C, int fN, int M, int k, int total) {
    int t = blockIdx.x * blockDim.x + threadIdx.x;
    if (t >= total) return;
    int m = t % M; int c = (t / M) % C; int b = t / (M * C);
    const int* ip = idx + ((size_t)b * M + m) * k;
    float mx = -3.4e38f;
    for (int i = 0; i < k; ++i) mx = fmaxf(mx, feat[((size_t)b * C + c) * fN + ip[i]]);
    out[t] = mx;
}

__global__ void rowmax_kernel(const float* __restrict__ src, float* __restrict__ dst,
                              int O, int Nc, int dstStride, int dstOff, int total) {
    int t = blockIdx.x * blockDim.x + threadIdx.x;
    if (t >= total) return;
    int o = t % O; int b = t / O;
    const float* p = src + ((size_t)b * O + o) * Nc;
    float m = -3.4e38f;
    for (int n = 0; n < Nc; ++n) m = fmaxf(m, p[n]);
    dst[(size_t)b * dstStride + dstOff + o] = m;
}

__global__ void fc_kernel(const float* __restrict__ X, const float* __restrict__ W,
                          const float* __restrict__ bias, float* __restrict__ Y,
                          int IN, int OUT, int total) {
    int t = blockIdx.x * blockDim.x + threadIdx.x;
    if (t >= total) return;
    int o = t % OUT; int b = t / OUT;
    float s = bias ? bias[o] : 0.f;
    const float* xr = X + (size_t)b * IN;
    const float* wr = W + (size_t)o * IN;
    for (int j = 0; j < IN; ++j) s += xr[j] * wr[j];
    Y[t] = s;
}

__global__ void bnrow_kernel(float* __restrict__ Y, const float* __restrict__ g,
                             const float* __restrict__ bb, int Bn, int F) {
    int f = blockIdx.x * blockDim.x + threadIdx.x;
    if (f >= F) return;
    float s = 0.f, q = 0.f;
    for (int b = 0; b < Bn; ++b) { float v = Y[(size_t)b * F + f]; s += v; q += v * v; }
    float mean = s / Bn;
    float var  = q / Bn - mean * mean;
    float sc   = g[f] * rsqrtf(var + EPSV);
    for (int b = 0; b < Bn; ++b) {
        float v = (Y[(size_t)b * F + f] - mean) * sc + bb[f];
        Y[(size_t)b * F + f] = leaky(v);
    }
}

// ---------------------------------------------------------------------------
extern "C" void kernel_launch(void* const* d_in, const int* in_sizes, int n_in,
                              void* d_out, int out_size, void* d_ws, size_t ws_size,
                              hipStream_t stream) {
    (void)in_sizes; (void)n_in; (void)out_size; (void)ws_size;
    const int Bn = 8, N = 2048;

    const float* X   = (const float*)d_in[0];
    const float *W1 = (const float*)d_in[1],  *G1 = (const float*)d_in[2],  *Bb1 = (const float*)d_in[3];
    const float *W2 = (const float*)d_in[4],  *G2 = (const float*)d_in[5],  *Bb2 = (const float*)d_in[6];
    const float *W2m= (const float*)d_in[7],  *G2m= (const float*)d_in[8],  *Bb2m= (const float*)d_in[9];
    const float *W2p= (const float*)d_in[10], *G2p= (const float*)d_in[11], *Bb2p= (const float*)d_in[12];
    const float *W3 = (const float*)d_in[13], *G3 = (const float*)d_in[14], *Bb3 = (const float*)d_in[15];
    const float *W4 = (const float*)d_in[16], *G4 = (const float*)d_in[17], *Bb4 = (const float*)d_in[18];
    const float *W4m= (const float*)d_in[19], *G4m= (const float*)d_in[20], *Bb4m= (const float*)d_in[21];
    const float *W4p= (const float*)d_in[22], *G4p= (const float*)d_in[23], *Bb4p= (const float*)d_in[24];
    const float *W5 = (const float*)d_in[25], *G5 = (const float*)d_in[26], *Bb5 = (const float*)d_in[27];
    const float *W6 = (const float*)d_in[28], *G6 = (const float*)d_in[29], *Bb6 = (const float*)d_in[30];
    const float *W6m= (const float*)d_in[31], *G6m= (const float*)d_in[32], *Bb6m= (const float*)d_in[33];
    const float *W6p= (const float*)d_in[34], *G6p= (const float*)d_in[35], *Bb6p= (const float*)d_in[36];
    const float *W7 = (const float*)d_in[37], *G7 = (const float*)d_in[38], *Bb7 = (const float*)d_in[39];
    const float *W8 = (const float*)d_in[40], *G8 = (const float*)d_in[41], *Bb8 = (const float*)d_in[42];
    const float *Wl1= (const float*)d_in[43];
    const float *G9 = (const float*)d_in[44], *B9 = (const float*)d_in[45];
    const float *Wl2= (const float*)d_in[46], *Bl2= (const float*)d_in[47];
    const float *G10= (const float*)d_in[48], *B10= (const float*)d_in[49];
    const float *Wl3= (const float*)d_in[50], *Bl3= (const float*)d_in[51];

    // workspace carve
    char* base = (char*)d_ws;
    size_t off = 0;
    auto alloc = [&](size_t bytes) -> char* {
        char* p = base + off;
        off = (off + bytes + 255) & ~(size_t)255;
        return p;
    };
    int*   idxb = (int*)  alloc((size_t)Bn * N * 32 * 4);
    float* x1   = (float*)alloc((size_t)Bn * 64  * 2048 * 4);
    float* x2   = (float*)alloc((size_t)Bn * 64  * 2048 * 4);
    float* xt1  = (float*)alloc((size_t)Bn * 512 * 2048 * 4);
    float* agg1 = (float*)alloc((size_t)Bn * 64  * 512  * 4);
    float* xp1  = (float*)alloc((size_t)Bn * 128 * 512  * 4);
    float* x3   = (float*)alloc((size_t)Bn * 128 * 512  * 4);
    float* x4   = (float*)alloc((size_t)Bn * 128 * 512  * 4);
    float* xt2  = (float*)alloc((size_t)Bn * 512 * 512  * 4);
    float* agg2 = (float*)alloc((size_t)Bn * 128 * 128  * 4);
    float* xp2  = (float*)alloc((size_t)Bn * 256 * 128  * 4);
    float* x5   = (float*)alloc((size_t)Bn * 256 * 128  * 4);
    float* x6   = (float*)alloc((size_t)Bn * 256 * 128  * 4);
    float* xt3  = (float*)alloc((size_t)Bn * 512 * 128  * 4);
    float* agg3 = (float*)alloc((size_t)Bn * 256 * 32   * 4);
    float* xp3  = (float*)alloc((size_t)Bn * 512 * 32   * 4);
    float* x7   = (float*)alloc((size_t)Bn * 512 * 32   * 4);
    float* xt4  = (float*)alloc((size_t)Bn * 512 * 32   * 4);
    float* feats= (float*)alloc((size_t)Bn * 2048 * 4);
    float* y1   = (float*)alloc((size_t)Bn * 512 * 4);
    float* y2   = (float*)alloc((size_t)Bn * 256 * 4);
    float* stats= (float*)alloc(2048 * 4);                         // chsum | chsq
    _Float16* xT16 = (_Float16*)alloc((size_t)Bn * 2048 * 128 * 2);
    _Float16* Wp   = (_Float16*)alloc((size_t)1024 * 1024 * 2);

    auto run_knn = [&](const float* q, int qN, int M, const float* ref, int rN,
                       int Nref, int C, int k) {
        size_t sm = (size_t)(Nref + C + 128) * 4 + 128 * 4;
        knn_kernel<<<dim3(M, Bn), 128, sm, stream>>>(q, ref, C, qN, rN, M, Nref, k,
                                                     C * qN, C * rN, idxb);
    };

    auto run_edge = [&](const float* xin, int Cin, int Nn, int Kk,
                        const float* Wf, const float* gf, const float* bf,
                        int O, const float* res, float* out) {
        int Cpad2 = ((2 * Cin + 31) / 32) * 32;
        int tw = (Cpad2 / 32) * (O / 16) * 32;
        packW_kernel<<<(tw + 127) / 128, 128, 0, stream>>>(Wf, Wp, O, 2 * Cin, Cpad2, tw);
        hipMemsetAsync(stats, 0, 2048 * 4, stream);
        int COLS = (Kk == 4) ? 16 : 32;
        int NT = COLS / Kk;
        dim3 grid(Nn / NT, Bn);
        size_t sm = (size_t)COLS * Cpad2 * 2 + COLS * 4;
        if (Kk == 32)
            edgeconv_gemm<32, 32><<<grid, 128, sm, stream>>>(xin, Cin, Nn, Cpad2, idxb, Wp, O, out, stats, stats + 1024);
        else if (Kk == 16)
            edgeconv_gemm<16, 32><<<grid, 128, sm, stream>>>(xin, Cin, Nn, Cpad2, idxb, Wp, O, out, stats, stats + 1024);
        else if (Kk == 8)
            edgeconv_gemm<8, 32><<<grid, 128, sm, stream>>>(xin, Cin, Nn, Cpad2, idxb, Wp, O, out, stats, stats + 1024);
        else
            edgeconv_gemm<4, 16><<<grid, 128, sm, stream>>>(xin, Cin, Nn, Cpad2, idxb, Wp, O, out, stats, stats + 1024);
        int tot = Bn * O * Nn;
        float invc = 1.f / ((float)Bn * Nn * Kk);
        finalize_kernel<<<(tot + 255) / 256, 256, 0, stream>>>(out, out, stats, stats + 1024,
                                                               gf, bf, res, O, Nn, invc, tot);
    };

    auto run_c1 = [&](const float* s1, int C1, int s1N, const float* s2, int C2, int Nc,
                      const float* Wf, const float* gf, const float* bf, int O, float* out) {
        int Ctot = C1 + C2;
        int Cpad = ((Ctot + 31) / 32) * 32;
        int tp = Bn * Nc * Cpad;
        packX_kernel<<<(tp + 255) / 256, 256, 0, stream>>>(s1, C1, s1N, s2, C2, xT16, Nc, Cpad, tp);
        int tw = (Cpad / 32) * (O / 16) * 32;
        packW_kernel<<<(tw + 127) / 128, 128, 0, stream>>>(Wf, Wp, O, Ctot, Cpad, tw);
        hipMemsetAsync(stats, 0, 2048 * 4, stream);
        c1_gemm<<<dim3(Nc / 16, Bn), 128, 0, stream>>>(xT16, Wp, out, stats, stats + 1024, Cpad, O, Nc);
        int tot = Bn * O * Nc;
        finalize_kernel<<<(tot + 255) / 256, 256, 0, stream>>>(out, out, stats, stats + 1024,
                                                               gf, bf, nullptr, O, Nc,
                                                               1.f / ((float)Bn * Nc), tot);
    };

    auto run_agg = [&](int M, int Nref, int k, const float* feat, int C, int fN, float* out) {
        run_knn(X, 2048, M, X, 2048, Nref, 3, k);   // coords are always first M/Nref cols of xyz
        int tot = Bn * C * M;
        gather_max_kernel<<<(tot + 255) / 256, 256, 0, stream>>>(feat, idxb, out, C, fN, M, k, tot);
    };

    // ---- stage graph ----
    run_knn(X, 2048, 2048, X, 2048, 2048, 3, 32);
    run_edge(X, 3, 2048, 32, W1, G1, Bb1, 64, nullptr, x1);

    run_knn(x1, 2048, 2048, x1, 2048, 2048, 64, 32);
    run_edge(x1, 64, 2048, 32, W2, G2, Bb2, 64, nullptr, x2);

    run_c1(x2, 64, 2048, nullptr, 0, 2048, W2m, G2m, Bb2m, 512, xt1);

    run_agg(512, 2048, 32, x2, 64, 2048, agg1);
    run_c1(x2, 64, 2048, agg1, 64, 512, W2p, G2p, Bb2p, 128, xp1);

    run_knn(xp1, 512, 512, xp1, 512, 512, 128, 16);
    run_edge(xp1, 128, 512, 16, W3, G3, Bb3, 128, nullptr, x3);

    run_knn(x3, 512, 512, x3, 512, 512, 128, 16);
    run_edge(x3, 128, 512, 16, W4, G4, Bb4, 128, xp1, x4);

    run_c1(x4, 128, 512, nullptr, 0, 512, W4m, G4m, Bb4m, 512, xt2);

    run_agg(128, 512, 16, x4, 128, 512, agg2);
    run_c1(x4, 128, 512, agg2, 128, 128, W4p, G4p, Bb4p, 256, xp2);

    run_knn(xp2, 128, 128, xp2, 128, 128, 256, 8);
    run_edge(xp2, 256, 128, 8, W5, G5, Bb5, 256, nullptr, x5);

    run_knn(x5, 128, 128, x5, 128, 128, 256, 8);
    run_edge(x5, 256, 128, 8, W6, G6, Bb6, 256, xp2, x6);

    run_c1(x6, 256, 128, nullptr, 0, 128, W6m, G6m, Bb6m, 512, xt3);

    run_agg(32, 128, 8, x6, 256, 128, agg3);
    run_c1(x6, 256, 128, agg3, 256, 32, W6p, G6p, Bb6p, 512, xp3);

    run_knn(xp3, 32, 32, xp3, 32, 32, 512, 4);
    run_edge(xp3, 512, 32, 4, W7, G7, Bb7, 512, nullptr, x7);

    run_knn(x7, 32, 32, x7, 32, 32, 512, 4);
    run_edge(x7, 512, 32, 4, W8, G8, Bb8, 512, xp3, xt4);

    rowmax_kernel<<<(Bn * 512 + 255) / 256, 256, 0, stream>>>(xt1, feats, 512, 2048, 2048, 0,    Bn * 512);
    rowmax_kernel<<<(Bn * 512 + 255) / 256, 256, 0, stream>>>(xt2, feats, 512, 512,  2048, 512,  Bn * 512);
    rowmax_kernel<<<(Bn * 512 + 255) / 256, 256, 0, stream>>>(xt3, feats, 512, 128,  2048, 1024, Bn * 512);
    rowmax_kernel<<<(Bn * 512 + 255) / 256, 256, 0, stream>>>(xt4, feats, 512, 32,   2048, 1536, Bn * 512);

    fc_kernel<<<(Bn * 512 + 255) / 256, 256, 0, stream>>>(feats, Wl1, nullptr, y1, 2048, 512, Bn * 512);
    bnrow_kernel<<<(512 + 127) / 128, 128, 0, stream>>>(y1, G9, B9, Bn, 512);
    fc_kernel<<<(Bn * 256 + 255) / 256, 256, 0, stream>>>(y1, Wl2, Bl2, y2, 512, 256, Bn * 256);
    bnrow_kernel<<<(256 + 127) / 128, 128, 0, stream>>>(y2, G10, B10, Bn, 256);
    fc_kernel<<<(Bn * 40 + 255) / 256, 256, 0, stream>>>(y2, Wl3, Bl3, (float*)d_out, 256, 40, Bn * 40);
}
